// STTNet_30236569764578
// MI455X (gfx1250) — matrix operand order repaired and checked by reading, not compiled, not run on hardware
//
#include <hip/hip_runtime.h>

// ---------------------------------------------------------------------------
// CDNA5 (gfx1250, wave32) implementation of the STTNet reference.
// All GEMM-shaped compute uses v_wmma_f32_16x16x32_f16 (f16 in, f32 acc),
// with 2x2 register blocking (wave = 32x32 output tile, 4 accumulators) to
// halve VMEM loads per WMMA.
// ---------------------------------------------------------------------------

typedef __attribute__((ext_vector_type(16))) _Float16 v16h;
typedef __attribute__((ext_vector_type(8)))  _Float16 v8h;
typedef __attribute__((ext_vector_type(8)))  float    v8f;

union HF16 { v16h v; v8h h[2]; };

__device__ __forceinline__ v8f wmma16(v16h a, v16h b, v8f c) {
  return __builtin_amdgcn_wmma_f32_16x16x32_f16(false, a, false, b, (short)0, c,
                                                false, false);
}

__device__ __forceinline__ float act_f(float v, int act) {
  if (act == 1) return v > 0.f ? v : 0.01f * v;        // leaky relu 0.01
  if (act == 2) return 1.f / (1.f + __expf(-v));       // sigmoid
  return v;
}

// ---------------------------------------------------------------------------
// WMMA GEMM, 2x2 blocked:  C[M,N] = A[M,K](f16,row) * Bt[N,K](f16,row)^T
// wave -> 32x32 tile (4 accs); M%32==0, N%32==0, K%32==0.
// block = 256 (8 waves): wave w -> mt = bx*4 + (w>>1), nt = by*2 + (w&1).
// epilogue: v = acc*scale[n] + shift[n]; act; optional f32 [M,N] / f16 [M,N].
// ---------------------------------------------------------------------------
__global__ void k_gemm(const _Float16* __restrict__ A, const _Float16* __restrict__ Bt,
                       const float* __restrict__ scale, const float* __restrict__ shift,
                       int act, float* __restrict__ Cf, _Float16* __restrict__ Ch,
                       int M, int N, int K) {
  int lane = threadIdx.x & 31;
  int wave = threadIdx.x >> 5;
  int mt = blockIdx.x * 4 + (wave >> 1);
  int nt = blockIdx.y * 2 + (wave & 1);
  if (mt * 32 >= M || nt * 32 >= N) return;       // wave-uniform
  int m0 = mt * 32, n0 = nt * 32;
  int kbA = (lane & 16) ? 16 : 0;                 // contiguous k, matches B
  int kbB = (lane & 16) ? 16 : 0;                 // contiguous k, matches A
  const _Float16* a0p = A + (size_t)(m0 + (lane & 15)) * K;
  const _Float16* a1p = a0p + (size_t)16 * K;
  const _Float16* b0p = Bt + (size_t)(n0 + (lane & 15)) * K;
  const _Float16* b1p = b0p + (size_t)16 * K;
  v8f acc00 = {}, acc01 = {}, acc10 = {}, acc11 = {};
  for (int k0 = 0; k0 < K; k0 += 32) {
    HF16 a0, a1, b0, b1;
    a0.h[0] = *(const v8h*)(a0p + k0 + kbA);
    a0.h[1] = *(const v8h*)(a0p + k0 + kbA + 8);
    a1.h[0] = *(const v8h*)(a1p + k0 + kbA);
    a1.h[1] = *(const v8h*)(a1p + k0 + kbA + 8);
    b0.v    = *(const v16h*)(b0p + k0 + kbB);
    b1.v    = *(const v16h*)(b1p + k0 + kbB);
    acc00 = wmma16(a0.v, b0.v, acc00);
    acc01 = wmma16(a0.v, b1.v, acc01);
    acc10 = wmma16(a1.v, b0.v, acc10);
    acc11 = wmma16(a1.v, b1.v, acc11);
  }
  int na = n0 + (lane & 15), nb = na + 16;
  float sca = scale ? scale[na] : 1.f, sha = shift ? shift[na] : 0.f;
  float scb = scale ? scale[nb] : 1.f, shb = shift ? shift[nb] : 0.f;
  int mb = m0 + ((lane & 16) ? 8 : 0);            // C layout: vgpr r -> M=mb+r
  for (int r = 0; r < 8; ++r) {
    size_t ma = (size_t)(mb + r), m2 = ma + 16;
    float v00 = act_f(acc00[r] * sca + sha, act);
    float v01 = act_f(acc01[r] * scb + shb, act);
    float v10 = act_f(acc10[r] * sca + sha, act);
    float v11 = act_f(acc11[r] * scb + shb, act);
    if (Cf) {
      Cf[ma * N + na] = v00; Cf[ma * N + nb] = v01;
      Cf[m2 * N + na] = v10; Cf[m2 * N + nb] = v11;
    }
    if (Ch) {
      Ch[ma * N + na] = (_Float16)v00; Ch[ma * N + nb] = (_Float16)v01;
      Ch[m2 * N + na] = (_Float16)v10; Ch[m2 * N + nb] = (_Float16)v11;
    }
  }
}

// ---------------------------------------------------------------------------
// 3x3 SAME conv, implicit GEMM, 2x2 blocked.  in: NHWC f16 [B,H,W,Cin]
// (Cin%32==0, W%32==0), wt: [CoutPad][9][Cin] f16 (CoutPad=coutTiles*16).
// Wave = 32 pixels (one row) x 32 Cout (2nd cout tile optional, uniform).
// epilogue: scale/shift per cout (< cvalid) + act; NHWC f16 (stride ohc)
// and/or NCHW f32 (first ofc channels).
// ---------------------------------------------------------------------------
__global__ void k_conv3(const _Float16* __restrict__ in, const _Float16* __restrict__ wt,
                        const float* __restrict__ scale, const float* __restrict__ shift,
                        int act, _Float16* __restrict__ outH, int ohc,
                        float* __restrict__ outF, int ofc,
                        int Bn, int H, int W, int Cin, int coutTiles, int cvalid) {
  int lane = threadIdx.x & 31;
  int wave = threadIdx.x >> 5;
  int ppRow = W >> 5;                             // 32-pixel tiles per row
  int pixPairs = Bn * H * ppRow;
  int coutPairs = (coutTiles + 1) >> 1;
  int pp = blockIdx.x * 4 + (wave >> 1);
  int cp = blockIdx.y * 2 + (wave & 1);
  if (pp >= pixPairs || cp >= coutPairs) return;  // wave-uniform
  bool hasN2 = (cp * 2 + 1) < coutTiles;          // wave-uniform
  int b = pp / (H * ppRow);
  int rem = pp - b * H * ppRow;
  int y = rem / ppRow;
  int x0 = (rem % ppRow) << 5;
  int xa = x0 + (lane & 15), xb = xa + 16;        // A rows = pixels
  int kbA = (lane & 16) ? 16 : 0;
  int kbB = (lane & 16) ? 16 : 0;
  int cwa = cp * 32 + (lane & 15);                // B rows = couts
  const _Float16* wrA = wt + (size_t)cwa * 9 * Cin;
  const _Float16* wrB = wt + (size_t)(cwa + 16) * 9 * Cin;
  v8h z8 = {};
  v8f acc00 = {}, acc01 = {}, acc10 = {}, acc11 = {};
  for (int tap = 0; tap < 9; ++tap) {
    int dy = tap / 3 - 1, dx = tap % 3 - 1;
    int yy = y + dy;
    int xxa = xa + dx, xxb = xb + dx;
    bool rowOk = (unsigned)yy < (unsigned)H;
    bool va = rowOk && ((unsigned)xxa < (unsigned)W);
    bool vb = rowOk && ((unsigned)xxb < (unsigned)W);
    const _Float16* apA = in + ((size_t)(b * H + yy) * W + xxa) * Cin;
    const _Float16* apB = in + ((size_t)(b * H + yy) * W + xxb) * Cin;
    for (int c0 = 0; c0 < Cin; c0 += 32) {
      HF16 a0, a1, b0, b1;
      if (va) {
        a0.h[0] = *(const v8h*)(apA + c0 + kbA);
        a0.h[1] = *(const v8h*)(apA + c0 + kbA + 8);
      } else { a0.h[0] = z8; a0.h[1] = z8; }
      if (vb) {
        a1.h[0] = *(const v8h*)(apB + c0 + kbA);
        a1.h[1] = *(const v8h*)(apB + c0 + kbA + 8);
      } else { a1.h[0] = z8; a1.h[1] = z8; }
      b0.v = *(const v16h*)(wrA + (size_t)tap * Cin + c0 + kbB);
      acc00 = wmma16(a0.v, b0.v, acc00);
      acc10 = wmma16(a1.v, b0.v, acc10);
      if (hasN2) {
        b1.v = *(const v16h*)(wrB + (size_t)tap * Cin + c0 + kbB);
        acc01 = wmma16(a0.v, b1.v, acc01);
        acc11 = wmma16(a1.v, b1.v, acc11);
      }
    }
  }
  int na = cwa, nb = cwa + 16;
  float sca = (scale && na < cvalid) ? scale[na] : 1.f;
  float sha = (shift && na < cvalid) ? shift[na] : 0.f;
  float scb = (scale && hasN2 && nb < cvalid) ? scale[nb] : 1.f;
  float shb = (shift && hasN2 && nb < cvalid) ? shift[nb] : 0.f;
  int mb = (lane & 16) ? 8 : 0;
  for (int r = 0; r < 8; ++r) {
    int xpa = x0 + mb + r, xpb = xpa + 16;
    size_t pa = (size_t)(b * H + y) * W + xpa;
    size_t pb = pa + 16;
    if (na < cvalid) {
      float v00 = act_f(acc00[r] * sca + sha, act);
      float v10 = act_f(acc10[r] * sca + sha, act);
      if (outH) { outH[pa * ohc + na] = (_Float16)v00; outH[pb * ohc + na] = (_Float16)v10; }
      if (outF && na < ofc) {
        outF[((size_t)(b * ofc + na) * H + y) * W + xpa] = v00;
        outF[((size_t)(b * ofc + na) * H + y) * W + xpb] = v10;
      }
    }
    if (hasN2 && nb < cvalid) {
      float v01 = act_f(acc01[r] * scb + shb, act);
      float v11 = act_f(acc11[r] * scb + shb, act);
      if (outH) { outH[pa * ohc + nb] = (_Float16)v01; outH[pb * ohc + nb] = (_Float16)v11; }
      if (outF && nb < ofc) {
        outF[((size_t)(b * ofc + nb) * H + y) * W + xpa] = v01;
        outF[((size_t)(b * ofc + nb) * H + y) * W + xpb] = v11;
      }
    }
  }
}

// ------------------------- elementwise / small kernels ---------------------
__global__ void k_bn_prep(const float* g, const float* be, const float* mu,
                          const float* var, const float* bias,
                          float* scale, float* shift, int n) {
  int i = blockIdx.x * 256 + threadIdx.x;
  if (i >= n) return;
  float s = g[i] * rsqrtf(var[i] + 1e-5f);
  scale[i] = s;
  shift[i] = be[i] + ((bias ? bias[i] : 0.f) - mu[i]) * s;
}

__global__ void k_cvt(const float* s, _Float16* d, int tot) {
  int i = blockIdx.x * 256 + threadIdx.x;
  if (i < tot) d[i] = (_Float16)s[i];
}

__global__ void k_cvtT(const float* s, _Float16* d, int R, int C) { // d[c*R+r]=s[r*C+c]
  int i = blockIdx.x * 256 + threadIdx.x;
  if (i >= R * C) return;
  int r = i / C, c = i - r * C;
  d[(size_t)c * R + r] = (_Float16)s[i];
}

__global__ void k_pack_conv_w(const float* w, _Float16* d, int O, int I, int Op, int Ip) {
  int i = blockIdx.x * 256 + threadIdx.x;
  int tot = Op * 9 * Ip;
  if (i >= tot) return;
  int ii = i % Ip, t = (i / Ip) % 9, o = i / (Ip * 9);
  float v = (o < O && ii < I) ? w[((size_t)o * I + ii) * 9 + t] : 0.f;
  d[i] = (_Float16)v;
}

__global__ void k_fill_h(_Float16* p, int tot) {
  int i = blockIdx.x * 256 + threadIdx.x;
  if (i < tot) p[i] = (_Float16)0.f;
}

__global__ void k_pack_x(const float* x, _Float16* d, int tot) { // NCHW->NHWC, C=1024,N=1024
  int i = blockIdx.x * 256 + threadIdx.x;
  if (i >= tot) return;
  int c = i & 1023, n = (i >> 10) & 1023, b = i >> 20;
  d[i] = (_Float16)x[((size_t)(b * 1024 + c) << 10) + n];
}

__global__ void k_build_pos(const float* eh, const float* ew, float* pos) {
  int i = blockIdx.x * 256 + threadIdx.x;     // 1024*64
  if (i >= 65536) return;
  int d = i & 63, n = i >> 6;
  pos[(size_t)n * 64 + d] = eh[d * 32 + (n >> 5)] + ew[d * 32 + (n & 31)];
}

__global__ void k_transpose(const float* xr, float* of, _Float16* oh, int tot) {
  int i = blockIdx.x * 256 + threadIdx.x;     // [B,64,1024] from NHWC [B,1024,64]
  if (i >= tot) return;
  int n = i & 1023, d = (i >> 10) & 63, b = i >> 16;
  float v = xr[((size_t)(b * 1024 + n) << 6) + d];
  of[i] = v; oh[i] = (_Float16)v;
}

__global__ void k_gather(const float* src, long bstride, const int* idx,
                         int rows, int cols, float* of, _Float16* oh, int tot) {
  int i = blockIdx.x * 256 + threadIdx.x;
  if (i >= tot) return;
  int c = i % cols, r = (i / cols) % rows, b = i / (cols * rows);
  float v = src[(long)b * bstride + (long)idx[b * rows + r] * cols + c];
  if (of) of[i] = v;
  if (oh) oh[i] = (_Float16)v;
}

__global__ void k_chanconv(const float* xr, const float* w, const float* bias,
                           const float* g, const float* be, const float* mu,
                           const float* var, float* hc) {
  int oc = blockIdx.x, b = blockIdx.y, t = threadIdx.x;
  __shared__ float red[256];
  float s = 0.f;
  for (int i = t; i < 65536; i += 256) {      // i = cin*1024 + n
    int cin = i >> 10, n = i & 1023;
    s += xr[((size_t)(b * 1024 + n) << 6) + cin] * w[(size_t)oc * 65536 + i];
  }
  red[t] = s; __syncthreads();
  for (int o = 128; o; o >>= 1) { if (t < o) red[t] += red[t + o]; __syncthreads(); }
  if (t == 0) {
    float v = red[0] + bias[oc];
    float sc = g[oc] * rsqrtf(var[oc] + 1e-5f);
    v = (v - mu[oc]) * sc + be[oc];
    hc[b * 8 + oc] = v > 0.f ? v : 0.01f * v;
  }
}

__global__ void k_probc(const float* hc, const float* w2, const float* b2, float* prob) {
  int c = threadIdx.x, b = blockIdx.x;
  if (c >= 64) return;
  float s = b2[c];
  for (int i = 0; i < 8; ++i) s += hc[b * 8 + i] * w2[c * 8 + i];
  prob[b * 64 + c] = 1.f / (1.f + __expf(-s));
}

__global__ void k_topk(const float* vals, int n, int k, int* idx) {
  extern __shared__ float sm[];
  float* v = sm; float* rv = sm + n; int* ri = (int*)(rv + 256);
  int b = blockIdx.x, t = threadIdx.x;
  for (int i = t; i < n; i += 256) v[i] = vals[b * n + i];
  __syncthreads();
  for (int it = 0; it < k; ++it) {
    float bm = -1e30f; int bi = 0;
    for (int i = t; i < n; i += 256) if (v[i] > bm) { bm = v[i]; bi = i; }
    rv[t] = bm; ri[t] = bi; __syncthreads();
    for (int o = 128; o; o >>= 1) {
      if (t < o && rv[t + o] > rv[t]) { rv[t] = rv[t + o]; ri[t] = ri[t + o]; }
      __syncthreads();
    }
    if (t == 0) { idx[b * k + it] = ri[0]; v[ri[0]] = -1e30f; }
    __syncthreads();
  }
}

// scores (q·(k+pe))*scale -> softmax over j -> O = att @ V, per (i,h,b)
__global__ void k_attn(const float* Q, const float* Kk, const float* V, const float* PE,
                       _Float16* O, int nq, int nk, int Dm, int Hh) {
  int i = blockIdx.x, h = blockIdx.y, b = blockIdx.z, t = threadIdx.x;
  int hd = Dm / Hh;
  float scl = rsqrtf((float)hd);
  extern __shared__ float sm[];
  float* att = sm; float* red = sm + nk;
  const float* q = Q + ((size_t)(b * nq + i) * Dm) + h * hd;
  float lmax = -1e30f;
  for (int j = t; j < nk; j += 256) {
    const float* kp = Kk + ((size_t)(b * nk + j) * Dm) + h * hd;
    const float* pp = PE + ((size_t)(b * nk + j) * Dm) + h * hd;
    float s = 0.f;
    for (int d = 0; d < hd; ++d) s += q[d] * (kp[d] + pp[d]);
    s *= scl; att[j] = s; lmax = fmaxf(lmax, s);
  }
  red[t] = lmax; __syncthreads();
  for (int o = 128; o; o >>= 1) { if (t < o) red[t] = fmaxf(red[t], red[t + o]); __syncthreads(); }
  float mx = red[0]; __syncthreads();
  float ls = 0.f;
  for (int j = t; j < nk; j += 256) { float e = __expf(att[j] - mx); att[j] = e; ls += e; }
  red[t] = ls; __syncthreads();
  for (int o = 128; o; o >>= 1) { if (t < o) red[t] += red[t + o]; __syncthreads(); }
  float inv = 1.f / red[0]; __syncthreads();
  for (int d = t; d < hd; d += 256) {
    float a = 0.f;
    for (int j = 0; j < nk; ++j) a += att[j] * V[((size_t)(b * nk + j) * Dm) + h * hd + d];
    O[((size_t)(b * nq + i) * Dm) + h * hd + d] = (_Float16)(a * inv);
  }
}

__global__ void k_ln_res(const float* O, const float* R, const float* g, const float* be,
                         float* outF, _Float16* outH, int Dm) {
  int row = blockIdx.x, t = threadIdx.x;
  __shared__ float red[256];
  const float* o = O + (size_t)row * Dm;
  const float* r = R + (size_t)row * Dm;
  float s = 0.f;
  for (int d = t; d < Dm; d += 256) s += o[d] + r[d];
  red[t] = s; __syncthreads();
  for (int k = 128; k; k >>= 1) { if (t < k) red[t] += red[t + k]; __syncthreads(); }
  float mean = red[0] / Dm; __syncthreads();
  float v = 0.f;
  for (int d = t; d < Dm; d += 256) { float x = o[d] + r[d] - mean; v += x * x; }
  red[t] = v; __syncthreads();
  for (int k = 128; k; k >>= 1) { if (t < k) red[t] += red[t + k]; __syncthreads(); }
  float inv = rsqrtf(red[0] / Dm + 1e-5f); __syncthreads();
  for (int d = t; d < Dm; d += 256) {
    float y = (o[d] + r[d] - mean) * inv * g[d] + be[d];
    if (outF) outF[(size_t)row * Dm + d] = y;
    if (outH) outH[(size_t)row * Dm + d] = (_Float16)y;
  }
}

__global__ void k_pack_cat(const _Float16* xrh, const float* ps, const float* pc,
                           const _Float16* tfs, const _Float16* tfc, _Float16* A, int tot) {
  int i = blockIdx.x * 256 + threadIdx.x;     // 2048*64
  if (i >= tot) return;
  int c = i & 63, row = i >> 6, b = row >> 10, n = row & 1023;
  float x = (float)xrh[(size_t)row * 64 + c];
  _Float16* a = A + (size_t)row * 320;
  a[c]       = (_Float16)x;
  a[64 + c]  = (_Float16)(x * ps[row]);
  a[128 + c] = (_Float16)(x * pc[b * 64 + c]);
  a[192 + c] = tfs[(size_t)row * 64 + c];
  a[256 + c] = tfc[((size_t)(b * 64 + c) << 10) + n];
}

__global__ void k_ps1(const _Float16* in, _Float16* out, int tot) { // [B,32,32,1024]->[B,128,128,64]
  int i = blockIdx.x * 256 + threadIdx.x;
  if (i >= tot) return;
  int c = i & 63, X = (i >> 6) & 127, Y = (i >> 13) & 127, b = i >> 20;
  out[i] = in[(((size_t)(b * 32 + (Y >> 2)) * 32 + (X >> 2)) << 10) + c * 16 + (Y & 3) * 4 + (X & 3)];
}

__global__ void k_ps2(const _Float16* in, _Float16* out, int tot) { // [B,128,128,256]->[B,512,512,32] (pad 16->32)
  int i = blockIdx.x * 256 + threadIdx.x;
  if (i >= tot) return;
  int c = i & 31, X = (i >> 5) & 511, Y = (i >> 14) & 511, b = i >> 23;
  _Float16 v = (_Float16)0.f;
  if (c < 16)
    v = in[((size_t)(b * 128 + (Y >> 2)) * 128 + (X >> 2)) * 256 + c * 16 + (Y & 3) * 4 + (X & 3)];
  out[i] = v;
}

// ---------------------------------------------------------------------------
extern "C" void kernel_launch(void* const* d_in, const int* in_sizes, int n_in,
                              void* d_out, int out_size, void* d_ws, size_t ws_size,
                              hipStream_t stream) {
  (void)in_sizes; (void)n_in; (void)out_size; (void)ws_size;
  auto F = [&](int i) { return (const float*)d_in[i]; };
  char* base = (char*)d_ws;
  size_t off = 0;
  auto alloc = [&](size_t bytes) -> void* {
    void* p = base + off; off += (bytes + 255) & ~(size_t)255; return p;
  };
  auto gb = [&](int tot) { return dim3((tot + 255) / 256); };

  // ---- scratch layout ----
  float*    scR  = (float*)alloc(64 * 4);    float* shR  = (float*)alloc(64 * 4);
  float*    scS  = (float*)alloc(16 * 4);    float* shS  = (float*)alloc(16 * 4);
  float*    scH  = (float*)alloc(1024 * 4);  float* shH  = (float*)alloc(1024 * 4);
  float*    scA1 = (float*)alloc(256 * 4);   float* shA1 = (float*)alloc(256 * 4);
  float*    scA2 = (float*)alloc(256 * 4);   float* shA2 = (float*)alloc(256 * 4);
  float*    scB1 = (float*)alloc(64 * 4);    float* shB1 = (float*)alloc(64 * 4);
  float*    scB2 = (float*)alloc(64 * 4);    float* shB2 = (float*)alloc(64 * 4);
  float*    scP  = (float*)alloc(64 * 4);    float* shP  = (float*)alloc(64 * 4);
  _Float16* Ax   = (_Float16*)alloc((size_t)2048 * 1024 * 2);
  _Float16* Wr   = (_Float16*)alloc(64 * 1024 * 2);
  float*    xrF  = (float*)alloc((size_t)2048 * 64 * 4);
  _Float16* xrH  = (_Float16*)alloc((size_t)2048 * 64 * 2);
  _Float16* Ws1  = (_Float16*)alloc(16 * 9 * 64 * 2);
  _Float16* Ws2  = (_Float16*)alloc(16 * 9 * 32 * 2);
  _Float16* hsH  = (_Float16*)alloc((size_t)2048 * 32 * 2);
  float*    probS= (float*)alloc(2048 * 4);
  float*    hc   = (float*)alloc(16 * 4);
  float*    probC= (float*)alloc(128 * 4);
  int*      idxS = (int*)alloc(512 * 4);
  int*      idxC = (int*)alloc(32 * 4);
  float*    pos  = (float*)alloc((size_t)1024 * 64 * 4);
  float*    XsF  = (float*)alloc((size_t)512 * 64 * 4);
  _Float16* XsH  = (_Float16*)alloc((size_t)512 * 64 * 2);
  float*    PosT = (float*)alloc((size_t)512 * 64 * 4);
  float*    Qb   = (float*)alloc((size_t)131072 * 4);
  float*    Kb   = (float*)alloc((size_t)131072 * 4);
  float*    Vb   = (float*)alloc((size_t)131072 * 4);
  _Float16* Ob   = (_Float16*)alloc((size_t)131072 * 2);
  float*    Pf   = (float*)alloc((size_t)131072 * 4);
  _Float16* EncS = (_Float16*)alloc((size_t)512 * 64 * 2);
  _Float16* TfS  = (_Float16*)alloc((size_t)2048 * 64 * 2);
  float*    XcF  = (float*)alloc((size_t)131072 * 4);
  _Float16* XcH  = (_Float16*)alloc((size_t)131072 * 2);
  float*    XtcF = (float*)alloc((size_t)32768 * 4);
  _Float16* XtcH = (_Float16*)alloc((size_t)32768 * 2);
  float*    PosC = (float*)alloc((size_t)32768 * 4);
  _Float16* EncC = (_Float16*)alloc((size_t)32768 * 2);
  _Float16* TfC  = (_Float16*)alloc((size_t)131072 * 2);
  _Float16* Wsm  = (_Float16*)alloc((size_t)8 * 64 * 64 * 2);      // 8 small attn mats
  _Float16* Wbg  = (_Float16*)alloc((size_t)8 * 1024 * 1024 * 2);  // 8 big attn mats
  _Float16* Acat = (_Float16*)alloc((size_t)2048 * 320 * 2);
  _Float16* Whd  = (_Float16*)alloc((size_t)1024 * 320 * 2);
  _Float16* Yh   = (_Float16*)alloc((size_t)2048 * 1024 * 2);
  _Float16* P1   = (_Float16*)alloc((size_t)2 * 128 * 128 * 64 * 2);
  _Float16* W1a  = (_Float16*)alloc((size_t)256 * 9 * 64 * 2);
  _Float16* C1   = (_Float16*)alloc((size_t)2 * 128 * 128 * 256 * 2);
  _Float16* W1b  = (_Float16*)alloc((size_t)256 * 9 * 256 * 2);
  _Float16* C1b  = (_Float16*)alloc((size_t)2 * 128 * 128 * 256 * 2);
  _Float16* P2   = (_Float16*)alloc((size_t)2 * 512 * 512 * 32 * 2);
  _Float16* W2a  = (_Float16*)alloc((size_t)64 * 9 * 32 * 2);
  _Float16* W2b  = (_Float16*)alloc((size_t)64 * 9 * 64 * 2);
  _Float16* Wp1  = (_Float16*)alloc((size_t)64 * 9 * 64 * 2);
  _Float16* Wp2  = (_Float16*)alloc((size_t)16 * 9 * 64 * 2);
  _Float16* C2   = (_Float16*)alloc((size_t)2 * 512 * 512 * 64 * 2);
  _Float16* C2b  = (_Float16*)alloc((size_t)2 * 512 * 512 * 64 * 2);

  auto gemm = [&](const _Float16* A, const _Float16* Bt, const float* sc,
                  const float* sh, int act, float* Cf, _Float16* Ch,
                  int M, int N, int K) {
    dim3 g((M / 32 + 3) / 4, (N / 32 + 1) / 2);
    k_gemm<<<g, 256, 0, stream>>>(A, Bt, sc, sh, act, Cf, Ch, M, N, K);
  };
  auto conv = [&](const _Float16* in, const _Float16* wt, const float* sc,
                  const float* sh, int act, _Float16* oh, int ohc, float* of,
                  int ofc, int B_, int H, int W, int Cin, int ct, int cv) {
    int pixPairs = B_ * H * (W >> 5);
    int coutPairs = (ct + 1) >> 1;
    dim3 g((pixPairs + 3) / 4, (coutPairs + 1) / 2);
    k_conv3<<<g, 256, 0, stream>>>(in, wt, sc, sh, act, oh, ohc, of, ofc,
                                   B_, H, W, Cin, ct, cv);
  };

  // ---- fold BN(+bias) -> scale/shift ----
  k_bn_prep<<<1, 256, 0, stream>>>(F(3),  F(4),  F(5),  F(6),  F(2),  scR,  shR,  64);
  k_bn_prep<<<1, 256, 0, stream>>>(F(9),  F(10), F(11), F(12), F(8),  scS,  shS,  16);
  k_bn_prep<<<4, 256, 0, stream>>>(F(56), F(57), F(58), F(59), F(55), scH,  shH,  1024);
  k_bn_prep<<<1, 256, 0, stream>>>(F(62), F(63), F(64), F(65), F(61), scA1, shA1, 256);
  k_bn_prep<<<1, 256, 0, stream>>>(F(68), F(69), F(70), F(71), F(67), scA2, shA2, 256);
  k_bn_prep<<<1, 256, 0, stream>>>(F(74), F(75), F(76), F(77), F(73), scB1, shB1, 64);
  k_bn_prep<<<1, 256, 0, stream>>>(F(80), F(81), F(82), F(83), F(79), scB2, shB2, 64);
  k_bn_prep<<<1, 256, 0, stream>>>(F(86), F(87), F(88), F(89), F(85), scP,  shP,  64);

  // ---- reduce 1x1: xr = lrelu(bn(x @ W)) ----
  k_pack_x<<<gb(2097152), 256, 0, stream>>>(F(0), Ax, 2097152);
  k_cvt<<<gb(65536), 256, 0, stream>>>(F(1), Wr, 65536);
  gemm(Ax, Wr, scR, shR, 1, xrF, xrH, 2048, 64, 1024);

  // ---- spatial prob map ----
  k_pack_conv_w<<<gb(16 * 9 * 64), 256, 0, stream>>>(F(7), Ws1, 16, 64, 16, 64);
  k_fill_h<<<gb(65536), 256, 0, stream>>>(hsH, 65536);
  conv(xrH, Ws1, scS, shS, 1, hsH, 32, nullptr, 0, 2, 32, 32, 64, 1, 16);
  k_pack_conv_w<<<gb(16 * 9 * 32), 256, 0, stream>>>(F(13), Ws2, 1, 16, 16, 32);
  conv(hsH, Ws2, nullptr, F(14), 2, nullptr, 0, probS, 1, 2, 32, 32, 32, 1, 1);

  // ---- channel prob map ----
  k_chanconv<<<dim3(8, 2), 256, 0, stream>>>(xrF, F(15), F(16), F(17), F(18), F(19), F(20), hc);
  k_probc<<<2, 64, 0, stream>>>(hc, F(21), F(22), probC);

  // ---- top-k ----
  k_topk<<<2, 256, (1024 + 512) * 4, stream>>>(probS, 1024, 256, idxS);
  k_topk<<<2, 256, (64 + 512) * 4, stream>>>(probC, 64, 16, idxC);

  // ---- spatial branch ----
  k_build_pos<<<gb(65536), 256, 0, stream>>>(F(23), F(24), pos);
  k_gather<<<gb(32768), 256, 0, stream>>>(xrF, 65536, idxS, 256, 64, XsF, XsH, 32768);
  k_gather<<<gb(32768), 256, 0, stream>>>(pos, 0, idxS, 256, 64, PosT, nullptr, 32768);
  for (int m = 0; m < 4; ++m) {   // enc_s wq/wk/wv/out_w (26..29), dec_s (33..36)
    k_cvtT<<<gb(4096), 256, 0, stream>>>(F(26 + m), Wsm + m * 4096, 64, 64);
    k_cvtT<<<gb(4096), 256, 0, stream>>>(F(33 + m), Wsm + (4 + m) * 4096, 64, 64);
  }
  // enc_s
  gemm(XsH, Wsm + 0 * 4096, nullptr, nullptr, 0, Qb, nullptr, 512, 64, 64);
  gemm(XsH, Wsm + 1 * 4096, nullptr, nullptr, 0, Kb, nullptr, 512, 64, 64);
  gemm(XsH, Wsm + 2 * 4096, nullptr, nullptr, 0, Vb, nullptr, 512, 64, 64);
  k_attn<<<dim3(256, 32, 2), 256, (256 + 256) * 4, stream>>>(Qb, Kb, Vb, PosT, Ob, 256, 256, 64, 32);
  gemm(Ob, Wsm + 3 * 4096, nullptr, F(30), 0, Pf, nullptr, 512, 64, 64);
  k_ln_res<<<512, 256, 0, stream>>>(Pf, XsF, F(31), F(32), nullptr, EncS, 64);
  // dec_s
  gemm(xrH,  Wsm + 4 * 4096, nullptr, nullptr, 0, Qb, nullptr, 2048, 64, 64);
  gemm(EncS, Wsm + 5 * 4096, nullptr, nullptr, 0, Kb, nullptr, 512, 64, 64);
  gemm(EncS, Wsm + 6 * 4096, nullptr, nullptr, 0, Vb, nullptr, 512, 64, 64);
  k_attn<<<dim3(1024, 32, 2), 256, (256 + 256) * 4, stream>>>(Qb, Kb, Vb, PosT, Ob, 1024, 256, 64, 32);
  gemm(Ob, Wsm + 7 * 4096, nullptr, F(37), 0, Pf, nullptr, 2048, 64, 64);
  k_ln_res<<<2048, 256, 0, stream>>>(Pf, xrF, F(38), F(39), nullptr, TfS, 64);

  // ---- channel branch ----
  k_transpose<<<gb(131072), 256, 0, stream>>>(xrF, XcF, XcH, 131072);
  k_gather<<<gb(32768), 256, 0, stream>>>(XcF, 65536, idxC, 16, 1024, XtcF, XtcH, 32768);
  k_gather<<<gb(32768), 256, 0, stream>>>(F(25), 0, idxC, 16, 1024, PosC, nullptr, 32768);
  for (int m = 0; m < 4; ++m) {   // enc_c (40..43), dec_c (47..50)
    k_cvtT<<<gb(1048576), 256, 0, stream>>>(F(40 + m), Wbg + (size_t)m * 1048576, 1024, 1024);
    k_cvtT<<<gb(1048576), 256, 0, stream>>>(F(47 + m), Wbg + (size_t)(4 + m) * 1048576, 1024, 1024);
  }
  // enc_c
  gemm(XtcH, Wbg + (size_t)0 * 1048576, nullptr, nullptr, 0, Qb, nullptr, 32, 1024, 1024);
  gemm(XtcH, Wbg + (size_t)1 * 1048576, nullptr, nullptr, 0, Kb, nullptr, 32, 1024, 1024);
  gemm(XtcH, Wbg + (size_t)2 * 1048576, nullptr, nullptr, 0, Vb, nullptr, 32, 1024, 1024);
  k_attn<<<dim3(16, 2, 2), 256, (16 + 256) * 4, stream>>>(Qb, Kb, Vb, PosC, Ob, 16, 16, 1024, 2);
  gemm(Ob, Wbg + (size_t)3 * 1048576, nullptr, F(44), 0, Pf, nullptr, 32, 1024, 1024);
  k_ln_res<<<32, 256, 0, stream>>>(Pf, XtcF, F(45), F(46), nullptr, EncC, 1024);
  // dec_c
  gemm(XcH,  Wbg + (size_t)4 * 1048576, nullptr, nullptr, 0, Qb, nullptr, 128, 1024, 1024);
  gemm(EncC, Wbg + (size_t)5 * 1048576, nullptr, nullptr, 0, Kb, nullptr, 32, 1024, 1024);
  gemm(EncC, Wbg + (size_t)6 * 1048576, nullptr, nullptr, 0, Vb, nullptr, 32, 1024, 1024);
  k_attn<<<dim3(64, 2, 2), 256, (16 + 256) * 4, stream>>>(Qb, Kb, Vb, PosC, Ob, 64, 16, 1024, 2);
  gemm(Ob, Wbg + (size_t)7 * 1048576, nullptr, F(51), 0, Pf, nullptr, 128, 1024, 1024);
  k_ln_res<<<128, 256, 0, stream>>>(Pf, XcF, F(52), F(53), nullptr, TfC, 1024);

  // ---- concat + head 1x1 + pixel-shuffle decoder ----
  k_pack_cat<<<gb(131072), 256, 0, stream>>>(xrH, probS, probC, TfS, TfC, Acat, 131072);
  k_cvt<<<gb(327680), 256, 0, stream>>>(F(54), Whd, 327680);
  gemm(Acat, Whd, scH, shH, 1, nullptr, Yh, 2048, 1024, 320);
  k_ps1<<<gb(2097152), 256, 0, stream>>>(Yh, P1, 2097152);

  k_pack_conv_w<<<gb(256 * 9 * 64), 256, 0, stream>>>(F(60), W1a, 256, 64, 256, 64);
  conv(P1, W1a, scA1, shA1, 1, C1, 256, nullptr, 0, 2, 128, 128, 64, 16, 256);
  k_pack_conv_w<<<gb(256 * 9 * 256), 256, 0, stream>>>(F(66), W1b, 256, 256, 256, 256);
  conv(C1, W1b, scA2, shA2, 1, C1b, 256, nullptr, 0, 2, 128, 128, 256, 16, 256);
  k_ps2<<<gb(16777216), 256, 0, stream>>>(C1b, P2, 16777216);

  k_pack_conv_w<<<gb(64 * 9 * 32), 256, 0, stream>>>(F(72), W2a, 64, 16, 64, 32);
  conv(P2, W2a, scB1, shB1, 1, C2, 64, nullptr, 0, 2, 512, 512, 32, 4, 64);
  k_pack_conv_w<<<gb(64 * 9 * 64), 256, 0, stream>>>(F(78), W2b, 64, 64, 64, 64);
  conv(C2, W2b, scB2, shB2, 1, C2b, 64, nullptr, 0, 2, 512, 512, 64, 4, 64);
  k_pack_conv_w<<<gb(64 * 9 * 64), 256, 0, stream>>>(F(84), Wp1, 64, 64, 64, 64);
  conv(C2b, Wp1, scP, shP, 1, C2, 64, nullptr, 0, 2, 512, 512, 64, 4, 64);
  k_pack_conv_w<<<gb(16 * 9 * 64), 256, 0, stream>>>(F(90), Wp2, 2, 64, 16, 64);
  conv(C2, Wp2, nullptr, F(91), 0, nullptr, 0, (float*)d_out, 2, 2, 512, 512, 64, 1, 2);
}
